// SNNModel_6545530159191
// MI455X (gfx1250) — matrix operand order repaired and checked by reading, not compiled, not run on hardware
//
#include <hip/hip_runtime.h>
#include <hip/hip_bf16.h>

// ---------------- types ----------------
typedef __attribute__((ext_vector_type(16))) __bf16       v16bf;
typedef __attribute__((ext_vector_type(8)))  __bf16       v8bf;
typedef __attribute__((ext_vector_type(4)))  __bf16       v4bf;
typedef __attribute__((ext_vector_type(8)))  float        v8f;
typedef __attribute__((ext_vector_type(4)))  float        f4v;   // native float4 (AS-safe)
typedef __attribute__((ext_vector_type(4)))  unsigned int u4v;   // native uint4  (AS-safe)
typedef __attribute__((__vector_size__(16))) int          vi4;   // int32x4 (builtin's pointee)

// address-space-qualified pointers: force global_load/global_store (not flat)
typedef const __attribute__((address_space(1))) u4v*   GcU4;
typedef const __attribute__((address_space(1))) f4v*   GcF4;
typedef __attribute__((address_space(1))) f4v*         GF4;
typedef const __attribute__((address_space(1))) float* GcF;
typedef __attribute__((address_space(1))) float*       GF;
typedef __attribute__((address_space(1))) vi4*         GAi4;  // async builtin src type
typedef __attribute__((address_space(3))) vi4*         LAi4;  // async builtin dst type

// CDNA5 async global->LDS path (guarded: falls back to register staging if absent)
#if defined(__AMDGCN__) && __has_builtin(__builtin_amdgcn_global_load_async_to_lds_b128)
#define USE_ASYNC 1
#else
#define USE_ASYNC 0
#endif

__device__ __forceinline__ void wait_asynccnt0() {
#if defined(__AMDGCN__)
#if __has_builtin(__builtin_amdgcn_s_wait_asynccnt)
    __builtin_amdgcn_s_wait_asynccnt(0);
#else
    asm volatile("s_wait_asynccnt 0" ::: "memory");
#endif
#endif
}

// Problem constants
#define BATCH 1024
#define DIN   4000
#define HID   2048
#define OUTN  1000
#define OPAD  1024
#define TSTEPS 10
#define CHUNK 400

// GEMM tiling
#define BLK_M 128
#define BLK_N 128
#define BLK_K 64
#define KP    72   // LDS row pitch in elements (144B): 16B-aligned, conflict-free b128 reads

// ---------------- convert x: f32 -> bf16 (same layout) ----------------
__global__ void cvt_x_kernel(const float* __restrict__ in, __bf16* __restrict__ out, int n4) {
    int i = blockIdx.x * blockDim.x + threadIdx.x;
    if (i >= n4) return;
    f4v f = ((GcF4)in)[i];
    v4bf o;
    o[0] = (__bf16)f[0]; o[1] = (__bf16)f[1]; o[2] = (__bf16)f[2]; o[3] = (__bf16)f[3];
    ((v4bf*)out)[i] = o;
}

// ---------------- transpose+convert: W[K][N] f32 -> Wt[Npad][K] bf16 ----------------
__global__ void transpose_cvt_kernel(const float* __restrict__ W, __bf16* __restrict__ Wt,
                                     int K, int N, int Npad) {
    __shared__ float tile[32][33];
    int kb = blockIdx.x * 32;
    int nb = blockIdx.y * 32;
#pragma unroll
    for (int r = 0; r < 4; ++r) {
        int row = kb + threadIdx.y + r * 8;   // k
        int col = nb + threadIdx.x;           // n
        float v = 0.0f;
        if (row < K && col < N) v = ((GcF)W)[(size_t)row * N + col];
        tile[threadIdx.y + r * 8][threadIdx.x] = v;
    }
    __syncthreads();
#pragma unroll
    for (int r = 0; r < 4; ++r) {
        int n = nb + threadIdx.y + r * 8;
        int k = kb + threadIdx.x;
        if (n < Npad && k < K)
            Wt[(size_t)n * K + k] = (__bf16)tile[threadIdx.x][threadIdx.y + r * 8];
    }
}

// ---------------- WMMA bf16 GEMM: C[M,N] (+)= A[M,K0:K1] * Bt[N,K0:K1]^T ----------------
// A row-major [M][lda] bf16, Bt row-major [N][ldb] bf16, C f32 [M][ldc].
// Grid (M/128, N/128); 256 threads = 8 waves (2 M x 4 N); wave tile 64x32.
// BLK_K=64 LDS slices, double-buffered; async DMA staged with a uniform fast path
// (no bounds logic) and a guarded zero-pad path only for the final partial slice.
__global__ __launch_bounds__(256)
void gemm_bf16_wmma(const __bf16* __restrict__ A, int lda,
                    const __bf16* __restrict__ Bt, int ldb,
                    float* __restrict__ C, int ldc,
                    const __bf16* __restrict__ zpad,   // >=16B of zeros (OOB source)
                    int K0, int K1, int accFlag) {
    __shared__ __bf16 Alds[2][BLK_M * KP];
    __shared__ __bf16 Blds[2][BLK_N * KP];

    const int tid  = threadIdx.x;
    const int lane = tid & 31;
    const int wid  = tid >> 5;
    const int wm   = wid >> 2;        // 0..1
    const int wn   = wid & 3;         // 0..3
    const int hf   = lane >> 4;
    const int l16  = lane & 15;

    const int blkM = blockIdx.x * BLK_M;
    const int blkN = blockIdx.y * BLK_N;

    const int srow = tid >> 3;        // 0..31 (+32*i)
    const int scol = (tid & 7) * 8;   // 0..56 step 8

    v8f acc[4][2];
#pragma unroll
    for (int i = 0; i < 4; ++i)
#pragma unroll
        for (int j = 0; j < 2; ++j)
#pragma unroll
            for (int e = 0; e < 8; ++e) acc[i][j][e] = 0.0f;

    // Per-thread global bases (advance by k at use site)
    const __bf16* aBase = A  + (size_t)(blkM + srow) * lda + scol;
    const __bf16* bBase = Bt + (size_t)(blkN + srow) * ldb + scol;
    const size_t aStep = (size_t)32 * lda;   // 32 rows between chunk groups
    const size_t bStep = (size_t)32 * ldb;

#if USE_ASYNC
    auto stageFast = [&](int k, int b) {      // uniform: whole slice in-range
#pragma unroll
        for (int i = 0; i < 4; ++i) {
            __builtin_amdgcn_global_load_async_to_lds_b128(
                (GAi4)(void*)(aBase + aStep * i + k),
                (LAi4)(void*)&Alds[b][(srow + 32 * i) * KP + scol], 0, 0);
            __builtin_amdgcn_global_load_async_to_lds_b128(
                (GAi4)(void*)(bBase + bStep * i + k),
                (LAi4)(void*)&Blds[b][(srow + 32 * i) * KP + scol], 0, 0);
        }
    };
    auto stageGuard = [&](int k, int b) {     // tail slice: OOB chunks read zero pad
        bool ok = (k + scol) < K1;            // chunk 8-aligned, K1 8-aligned
#pragma unroll
        for (int i = 0; i < 4; ++i) {
            __builtin_amdgcn_global_load_async_to_lds_b128(
                (GAi4)(void*)(ok ? (aBase + aStep * i + k) : zpad),
                (LAi4)(void*)&Alds[b][(srow + 32 * i) * KP + scol], 0, 0);
            __builtin_amdgcn_global_load_async_to_lds_b128(
                (GAi4)(void*)(ok ? (bBase + bStep * i + k) : zpad),
                (LAi4)(void*)&Blds[b][(srow + 32 * i) * KP + scol], 0, 0);
        }
    };
    auto stage = [&](int k, int b) {
        if (k + BLK_K <= K1) stageFast(k, b); // uniform branch
        else                 stageGuard(k, b);
    };
#else
    u4v aR[4], bR[4];
    auto loadG = [&](int k) {
        bool ok = (k + scol) < K1;
#pragma unroll
        for (int i = 0; i < 4; ++i) {
            aR[i] = *(GcU4)(ok ? (aBase + aStep * i + k) : zpad);
            bR[i] = *(GcU4)(ok ? (bBase + bStep * i + k) : zpad);
        }
    };
    auto storeLDS = [&](int b) {
#pragma unroll
        for (int i = 0; i < 4; ++i) {
            *(u4v*)(&Alds[b][(srow + 32 * i) * KP + scol]) = aR[i];
            *(u4v*)(&Blds[b][(srow + 32 * i) * KP + scol]) = bR[i];
        }
    };
#endif

    auto compute = [&](int b) {
#pragma unroll
        for (int ks = 0; ks < BLK_K; ks += 32) {
            v16bf af[4];
#pragma unroll
            for (int im = 0; im < 4; ++im) {
                const __bf16* p = &Alds[b][(wm * 64 + im * 16 + l16) * KP + ks];
                v8bf lo = *(const v8bf*)(p + 8 * hf);
                v8bf hi = *(const v8bf*)(p + 16 + 8 * hf);
                af[im] = __builtin_shufflevector(lo, hi, 0,1,2,3,4,5,6,7,8,9,10,11,12,13,14,15);
            }
            v16bf bfr[2];
#pragma unroll
            for (int jn = 0; jn < 2; ++jn) {
                const __bf16* p = &Blds[b][(wn * 32 + jn * 16 + l16) * KP + ks + 8 * hf];
                v8bf lo = *(const v8bf*)(p);
                v8bf hi = *(const v8bf*)(p + 16);
                bfr[jn] = __builtin_shufflevector(lo, hi, 0,1,2,3,4,5,6,7,8,9,10,11,12,13,14,15);
            }
#pragma unroll
            for (int im = 0; im < 4; ++im)
#pragma unroll
                for (int jn = 0; jn < 2; ++jn)
                    acc[im][jn] = __builtin_amdgcn_wmma_f32_16x16x32_bf16(
                        false, af[im], false, bfr[jn], (short)0, acc[im][jn], false, false);
        }
    };

    int buf = 0;
#if USE_ASYNC
    stage(K0, 0);
    for (int k = K0; k < K1; k += BLK_K) {
        wait_asynccnt0();                      // my slice loads for `buf` landed
        __syncthreads();                       // everyone's landed; prev reads of buf^1 done
        if (k + BLK_K < K1) stage(k + BLK_K, buf ^ 1);   // DMA next slice under compute
        compute(buf);
        buf ^= 1;
    }
#else
    loadG(K0);
    storeLDS(0);
    for (int k = K0; k < K1; k += BLK_K) {
        __syncthreads();                       // tile `buf` visible to all waves
        if (k + BLK_K < K1) loadG(k + BLK_K);  // issue next global loads early
        compute(buf);
        if (k + BLK_K < K1) storeLDS(buf ^ 1); // relay into other buffer post-compute
        buf ^= 1;
    }
#endif

    // Writeback: C frag VGPR r -> m = r + 8*half, n = lane&15
    GF Cg = (GF)C;
#pragma unroll
    for (int im = 0; im < 4; ++im)
#pragma unroll
        for (int jn = 0; jn < 2; ++jn) {
            int rowb = blkM + wm * 64 + im * 16 + 8 * hf;
            int col  = blkN + wn * 32 + jn * 16 + l16;
#pragma unroll
            for (int r = 0; r < 8; ++r) {
                size_t idx = (size_t)(rowb + r) * ldc + col;
                float val = acc[im][jn][r];
                if (accFlag) val += Cg[idx];
                Cg[idx] = val;
            }
        }
}

// ---------------- LIF layers 1/2 ----------------
__global__ void lif_kernel(const float* __restrict__ chg, const float* __restrict__ bias,
                           float* __restrict__ v, __bf16* __restrict__ s, int ncols) {
    int i = (blockIdx.x * blockDim.x + threadIdx.x) * 4;
    int col = i & (ncols - 1);
    f4v c4 = *(GcF4)(chg + i);
    f4v v4 = *(GcF4)(v + i);
    f4v b4 = *(GcF4)(bias + col);
    v4bf so;
#pragma unroll
    for (int j = 0; j < 4; ++j) {
        float vn = 0.5f * (v4[j] + c4[j] + b4[j]);
        bool sp = vn >= 1.0f;
        so[j] = sp ? (__bf16)1.0f : (__bf16)0.0f;
        v4[j] = sp ? 0.0f : vn;
    }
    *(GF4)(v + i) = v4;
    *(v4bf*)(s + i) = so;
}

// ---------------- LIF layer 3 ----------------
__global__ void lif3_kernel(const float* __restrict__ chg, const float* __restrict__ bias,
                            float* __restrict__ v, float* __restrict__ out) {
    int i = blockIdx.x * blockDim.x + threadIdx.x;   // BATCH*OPAD
    int b = i >> 10;
    int n = i & (OPAD - 1);
    float c = ((GcF)chg)[i] + (n < OUTN ? ((GcF)bias)[n] : 0.0f);
    float vn = 0.5f * (((GcF)v)[i] + c);
    bool sp = vn >= 1.0f;
    ((GF)v)[i] = sp ? 0.0f : vn;
    if (out != nullptr && n < OUTN) ((GF)out)[(size_t)b * OUTN + n] = sp ? 1.0f : 0.0f;
}

// ---------------- host launch ----------------
extern "C" void kernel_launch(void* const* d_in, const int* in_sizes, int n_in,
                              void* d_out, int out_size, void* d_ws, size_t ws_size,
                              hipStream_t stream) {
    const float* x  = (const float*)d_in[0];
    const float* W1 = (const float*)d_in[1];
    const float* b1 = (const float*)d_in[2];
    const float* W2 = (const float*)d_in[3];
    const float* b2 = (const float*)d_in[4];
    const float* W3 = (const float*)d_in[5];
    const float* b3 = (const float*)d_in[6];
    float* out = (float*)d_out;

    char* base = (char*)d_ws;
    size_t off = 0;
    auto carve = [&](size_t bytes) {
        char* p = base + off;
        off = (off + bytes + 255) & ~(size_t)255;
        return p;
    };
    __bf16* xbf  = (__bf16*)carve((size_t)BATCH * DIN * 2);
    __bf16* w1t  = (__bf16*)carve((size_t)HID * DIN * 2);
    __bf16* w2t  = (__bf16*)carve((size_t)HID * HID * 2);
    __bf16* w3t  = (__bf16*)carve((size_t)OPAD * HID * 2);
    float*  acc1 = (float*) carve((size_t)BATCH * HID * 4);
    float*  v1   = (float*) carve((size_t)BATCH * HID * 4);
    __bf16* s1   = (__bf16*)carve((size_t)BATCH * HID * 2);
    float*  c2   = (float*) carve((size_t)BATCH * HID * 4);
    float*  v2   = (float*) carve((size_t)BATCH * HID * 4);
    __bf16* s2   = (__bf16*)carve((size_t)BATCH * HID * 2);
    float*  c3   = (float*) carve((size_t)BATCH * OPAD * 4);
    float*  v3   = (float*) carve((size_t)BATCH * OPAD * 4);
    __bf16* zpad = (__bf16*)carve(256);
    (void)ws_size; (void)in_sizes; (void)n_in; (void)out_size;

    (void)hipMemsetAsync(acc1, 0, (size_t)BATCH * HID * 4, stream);
    (void)hipMemsetAsync(v1,   0, (size_t)BATCH * HID * 4, stream);
    (void)hipMemsetAsync(v2,   0, (size_t)BATCH * HID * 4, stream);
    (void)hipMemsetAsync(v3,   0, (size_t)BATCH * OPAD * 4, stream);
    (void)hipMemsetAsync(zpad, 0, 256, stream);

    cvt_x_kernel<<<(BATCH * DIN / 4 + 255) / 256, 256, 0, stream>>>(x, xbf, BATCH * DIN / 4);
    transpose_cvt_kernel<<<dim3(DIN / 32, HID / 32), dim3(32, 8), 0, stream>>>(W1, w1t, DIN, HID, HID);
    transpose_cvt_kernel<<<dim3(HID / 32, HID / 32), dim3(32, 8), 0, stream>>>(W2, w2t, HID, HID, HID);
    transpose_cvt_kernel<<<dim3(HID / 32, OPAD / 32), dim3(32, 8), 0, stream>>>(W3, w3t, HID, OUTN, OPAD);

    const int lifBlocks  = (BATCH * HID / 4 + 255) / 256;
    const int lif3Blocks = (BATCH * OPAD + 255) / 256;

    for (int t = 0; t < TSTEPS; ++t) {
        int K0 = t * CHUNK, K1 = (t + 1) * CHUNK;
        gemm_bf16_wmma<<<dim3(BATCH / BLK_M, HID / BLK_N), 256, 0, stream>>>(
            xbf, DIN, w1t, DIN, acc1, HID, zpad, K0, K1, /*acc=*/1);
        lif_kernel<<<lifBlocks, 256, 0, stream>>>(acc1, b1, v1, s1, HID);
        gemm_bf16_wmma<<<dim3(BATCH / BLK_M, HID / BLK_N), 256, 0, stream>>>(
            s1, HID, w2t, HID, c2, HID, zpad, 0, HID, /*acc=*/0);
        lif_kernel<<<lifBlocks, 256, 0, stream>>>(c2, b2, v2, s2, HID);
        gemm_bf16_wmma<<<dim3(BATCH / BLK_M, OPAD / BLK_N), 256, 0, stream>>>(
            s2, HID, w3t, HID, c3, OPAD, zpad, 0, HID, /*acc=*/0);
        lif3_kernel<<<lif3Blocks, 256, 0, stream>>>(c3, b3, v3, (t == TSTEPS - 1) ? out : nullptr);
    }
}
